// CrfDecoder_5033701671457
// MI455X (gfx1250) — matrix operand order repaired and checked by reading, not compiled, not run on hardware
//
#include <hip/hip_runtime.h>

typedef __attribute__((ext_vector_type(16))) _Float16 v16h;
typedef __attribute__((ext_vector_type(8)))  _Float16 v8h;
typedef __attribute__((ext_vector_type(8)))  float    v8f;
typedef __attribute__((ext_vector_type(4)))  float    v4f;

#define BB 512
#define SS 1024
#define LL 50
#define AST 68   // alpha f32 row stride (floats)  -> 272B, 16B aligned
#define EST 72   // exp(alpha) f16 row stride (halves) -> 144B, 16B aligned

#define LN2F 0.69314718055994531f

// ---------------------------------------------------------------------------
// Forward (denominator) kernel: 32 blocks x 128 threads.
// Block handles 16 batch rows; 4 waves each own one 16-wide N-tile.
// new_alpha[j] = m + ln( exp(alpha-m) @ exp(T) )[j] + logit[t,j]
// ---------------------------------------------------------------------------
__launch_bounds__(128)
__global__ void crf_forward(const float* __restrict__ logits,
                            const unsigned char* __restrict__ mask,
                            const float* __restrict__ trans,
                            const float* __restrict__ start_t,
                            const float* __restrict__ end_t,
                            float* __restrict__ den)
{
    __shared__ alignas(16) _Float16 eT[64 * 64];      // exp(T), n-major: eT[n*64+k]
    __shared__ alignas(16) float    alpha[16 * AST];  // pad cols hold -1e30
    __shared__ alignas(16) _Float16 aExp[16 * EST];   // exp(alpha - m), row-major
    __shared__ alignas(16) float    mrow[16];         // per-row max
    __shared__ alignas(16) float    mskf[16];         // mask[t] per row

    const int tid  = threadIdx.x;
    const int lane = tid & 31;
    const int wave = tid >> 5;            // 0..3  == N-tile index
    const int r0   = blockIdx.x * 16;     // batch row base

    // --- exp(T) into LDS (n-major, zero padded) -----------------------------
    for (int idx = tid; idx < 64 * 64; idx += 128) {
        const int n = idx >> 6, k = idx & 63;
        float v = 0.0f;
        if (n < LL && k < LL) v = __expf(trans[k * LL + n]);
        eT[idx] = (_Float16)v;
    }
    // --- alpha0 = start + logits[:,0,:] ------------------------------------
    {
        const int  r  = tid >> 3;
        const int  c0 = (tid & 7) * 8;
        const long b  = r0 + r;
        for (int c = c0; c < c0 + 8; ++c) {
            float v = -1e30f;
            if (c < LL) v = start_t[c] + logits[(b * SS + 0) * LL + c];
            alpha[r * AST + c] = v;
        }
    }
    __syncthreads();

    // --- B fragments (exp(T)) held in VGPRs for the whole scan --------------
    const int nb = wave;
    const int nl = lane & 15;
    const int hi = lane >> 4;
    v16h bf[2];
    {
        const int nn = nb * 16 + nl;
        #pragma unroll
        for (int kc = 0; kc < 2; ++kc) {
            const int kb = kc * 32 + hi * 16;
            v8h lo = *(const v8h*)&eT[nn * 64 + kb];
            v8h hh = *(const v8h*)&eT[nn * 64 + kb + 8];
            v16h f;
            #pragma unroll
            for (int i = 0; i < 8; ++i) { f[i] = lo[i]; f[i + 8] = hh[i]; }
            bf[kc] = f;
        }
    }

    const int  rA      = tid >> 3;        // phase-A row (0..15)
    const int  cA      = (tid & 7) * 8;   // phase-A col base
    const int  n       = nb * 16 + nl;    // output column of this lane
    const long rowbase = (long)(r0 + hi * 8) * SS;   // for writeback loads

    // --- this lane's 8 alpha cells, kept in registers across the scan -------
    float aold[8];
    if (n < LL) {
        #pragma unroll
        for (int v = 0; v < 8; ++v) aold[v] = alpha[(v + hi * 8) * AST + n];
    }

    #pragma unroll 1
    for (int t = 1; t < SS; ++t) {
        __syncthreads();  // previous step's alpha writes visible

        // ---- issue this step's logit loads early (hidden by phase A+WMMA) --
        float lgv[8];
        if (n < LL) {
            const float* lgp = logits + (rowbase + t) * LL + n;
            #pragma unroll
            for (int v = 0; v < 8; ++v) lgv[v] = lgp[(long)v * (SS * LL)];
        }
        if (tid < 16)
            mskf[tid] = (float)mask[(long)(r0 + tid) * SS + t];
        if (tid < 32 && t + 1 < SS) {
            const long b = r0 + (tid & 15);
            __builtin_prefetch(&logits[(b * SS + (t + 1)) * LL + (tid >> 4) * 32], 0, 3);
        }

        // ---- phase A: row max, exp, pack f16 -------------------------------
        v4f a0 = *(const v4f*)&alpha[rA * AST + cA];
        v4f a1 = *(const v4f*)&alpha[rA * AST + cA + 4];
        float mx = a0[0];
        #pragma unroll
        for (int i = 1; i < 4; ++i) mx = fmaxf(mx, a0[i]);
        #pragma unroll
        for (int i = 0; i < 4; ++i) mx = fmaxf(mx, a1[i]);
        mx = fmaxf(mx, __shfl_xor(mx, 1));
        mx = fmaxf(mx, __shfl_xor(mx, 2));
        mx = fmaxf(mx, __shfl_xor(mx, 4));   // all 8 lanes of row share max
        if ((tid & 7) == 0) mrow[rA] = mx;
        v8h h;
        #pragma unroll
        for (int i = 0; i < 4; ++i) {
            h[i]     = (_Float16)__expf(a0[i] - mx);
            h[i + 4] = (_Float16)__expf(a1[i] - mx);
        }
        *(v8h*)&aExp[rA * EST + cA] = h;
        __syncthreads();

        // ---- WMMA: C = exp(alpha-m) @ exp(T)   (K=64 as 2 chunks of 32) ----
        v16h af[2];
        #pragma unroll
        for (int kc = 0; kc < 2; ++kc) {
            const int kb = kc * 32 + hi * 16;
            v8h lo = *(const v8h*)&aExp[nl * EST + kb];
            v8h hh = *(const v8h*)&aExp[nl * EST + kb + 8];
            v16h f;
            #pragma unroll
            for (int i = 0; i < 8; ++i) { f[i] = lo[i]; f[i + 8] = hh[i]; }
            af[kc] = f;
        }
        v8f acc = {};
        acc = __builtin_amdgcn_wmma_f32_16x16x32_f16(false, af[0], false, bf[0],
                                                     (short)0, acc, false, false);
        acc = __builtin_amdgcn_wmma_f32_16x16x32_f16(false, af[1], false, bf[1],
                                                     (short)0, acc, false, false);

        // ---- writeback: branch-free, old alpha kept in registers -----------
        v4f mr0 = *(const v4f*)&mrow[hi * 8];
        v4f mr1 = *(const v4f*)&mrow[hi * 8 + 4];
        v4f mk0 = *(const v4f*)&mskf[hi * 8];
        v4f mk1 = *(const v4f*)&mskf[hi * 8 + 4];
        if (n < LL) {
            #pragma unroll
            for (int v = 0; v < 8; ++v) {
                const float mr = (v < 4) ? mr0[v] : mr1[v - 4];
                const float mm = (v < 4) ? mk0[v] : mk1[v - 4];
                float na = mr + LN2F * __log2f(acc[v]) + lgv[v];
                na = (mm != 0.0f) ? na : aold[v];     // mask==0 keeps old alpha
                aold[v] = na;
                alpha[(v + hi * 8) * AST + n] = na;
            }
        }
    }
    __syncthreads();

    // --- denominator: logsumexp(alpha + end_t) per row ----------------------
    {
        float vals[8];
        #pragma unroll
        for (int i = 0; i < 8; ++i) {
            const int c = cA + i;
            vals[i] = (c < LL) ? (alpha[rA * AST + c] + end_t[c]) : -1e30f;
        }
        float mx = vals[0];
        #pragma unroll
        for (int i = 1; i < 8; ++i) mx = fmaxf(mx, vals[i]);
        mx = fmaxf(mx, __shfl_xor(mx, 1));
        mx = fmaxf(mx, __shfl_xor(mx, 2));
        mx = fmaxf(mx, __shfl_xor(mx, 4));
        float s = 0.0f;
        #pragma unroll
        for (int i = 0; i < 8; ++i) s += __expf(vals[i] - mx);
        s += __shfl_xor(s, 1);
        s += __shfl_xor(s, 2);
        s += __shfl_xor(s, 4);
        if ((tid & 7) == 0) den[r0 + rA] = mx + LN2F * __log2f(s);
    }
}

// ---------------------------------------------------------------------------
// Numerator + final reduction: one thread per batch row.
// ---------------------------------------------------------------------------
__global__ void crf_numerator(const float* __restrict__ logits,
                              const int* __restrict__ tags,
                              const unsigned char* __restrict__ mask,
                              const float* __restrict__ trans,
                              const float* __restrict__ start_t,
                              const float* __restrict__ end_t,
                              const float* __restrict__ den,
                              float* __restrict__ out)
{
    const int b = blockIdx.x * blockDim.x + threadIdx.x;
    if (b >= BB) return;
    const int*           tg = tags + (long)b * SS;
    const unsigned char* mk = mask + (long)b * SS;
    const float*         lg = logits + (long)b * SS * LL;

    int   prev  = tg[0];
    float score = start_t[prev];
    int   cnt   = (int)mk[0];
    for (int t = 1; t < SS; ++t) {
        const int   cur = tg[t];
        const float mf  = (float)mk[t];
        score += trans[prev * LL + cur] * mf;                // transition t-1 -> t
        score += lg[(t - 1) * LL + prev] * (float)mk[t - 1]; // emit for t-1 (<= S-2)
        cnt  += (int)mk[t];
        prev  = cur;
    }
    const int li = cnt - 1;                 // last valid index
    const int lt = tg[li];
    score += end_t[lt];
    score += lg[(SS - 1) * LL + lt] * (float)mk[SS - 1];
    atomicAdd(out, score - den[b]);
}

__global__ void zero_out(float* out) { out[0] = 0.0f; }

// ---------------------------------------------------------------------------
extern "C" void kernel_launch(void* const* d_in, const int* in_sizes, int n_in,
                              void* d_out, int out_size, void* d_ws, size_t ws_size,
                              hipStream_t stream)
{
    const float*         logits  = (const float*)d_in[0];
    const int*           tags    = (const int*)d_in[1];
    const unsigned char* mask    = (const unsigned char*)d_in[2];
    const float*         trans   = (const float*)d_in[3];
    const float*         start_t = (const float*)d_in[4];
    const float*         end_t   = (const float*)d_in[5];
    float* out = (float*)d_out;
    float* den = (float*)d_ws;            // BB floats

    zero_out<<<1, 1, 0, stream>>>(out);
    crf_forward<<<BB / 16, 128, 0, stream>>>(logits, mask, trans, start_t, end_t, den);
    crf_numerator<<<(BB + 255) / 256, 256, 0, stream>>>(logits, tags, mask, trans,
                                                        start_t, end_t, den, out);
}
